// SlotPathC_44032004718732
// MI455X (gfx1250) — matrix operand order repaired and checked by reading, not compiled, not run on hardware
//
#include <hip/hip_runtime.h>
#include <math.h>

// ---------------- problem constants ----------------
#define DM   1024           // d_model
#define HH   512            // route hidden
#define SS   64             // num slots
#define TT   4096           // seq len
#define BB   4              // batch
#define BT   (BB*TT)        // 16384 tokens
#define TOPK 8

typedef __attribute__((ext_vector_type(8)))  __bf16 v8bf;
typedef __attribute__((ext_vector_type(16))) __bf16 v16bf;
typedef __attribute__((ext_vector_type(8)))  float  v8f;

// ---------------- helpers ----------------
__device__ inline v8bf cvt8(const float* __restrict__ p) {
    float4 a = *(const float4*)p;
    float4 b = *(const float4*)(p + 4);
    v8bf r;
    r[0]=(__bf16)a.x; r[1]=(__bf16)a.y; r[2]=(__bf16)a.z; r[3]=(__bf16)a.w;
    r[4]=(__bf16)b.x; r[5]=(__bf16)b.y; r[6]=(__bf16)b.z; r[7]=(__bf16)b.w;
    return r;
}
__device__ inline v16bf cat16(v8bf lo, v8bf hi) {
    return __builtin_shufflevector(lo, hi, 0,1,2,3,4,5,6,7,8,9,10,11,12,13,14,15);
}
__device__ inline v8f wmma_bf16(v16bf a, v16bf b, v8f c) {
    return __builtin_amdgcn_wmma_f32_16x16x32_bf16(false, a, false, b, (short)0, c, false, false);
}

// A-fragment loaders (per-lane: 8 contiguous K, per ISA 16-bit A layout)
__device__ inline v8bf loadA8(const float* __restrict__ A, const float* __restrict__ A2,
                              size_t row, int ldA, int Ksplit, int kb) {
    const float* p = (kb < Ksplit) ? (A + row * (size_t)ldA + kb) : (A2 + (kb - Ksplit));
    return cvt8(p);
}
__device__ inline v8bf loadA8(const __bf16* __restrict__ A, const float* __restrict__,
                              size_t row, int ldA, int, int kb) {
    return *(const v8bf*)(A + row * (size_t)ldA + kb);
}

// packed bf16 x8 -> one ds_store_b128
__device__ inline void storeW8(__bf16* dst, float4 w0, float4 w1) {
    union { __bf16 h[8]; uint4 u; } pk;
    pk.h[0]=(__bf16)w0.x; pk.h[1]=(__bf16)w0.y; pk.h[2]=(__bf16)w0.z; pk.h[3]=(__bf16)w0.w;
    pk.h[4]=(__bf16)w1.x; pk.h[5]=(__bf16)w1.y; pk.h[6]=(__bf16)w1.z; pk.h[7]=(__bf16)w1.w;
    *(uint4*)dst = pk.u;
}

// ---------------- generic WMMA GEMM:  C[M,N] = act(A[M,K] @ W[N,K]^T + bias) ----------------
// block = 256 thr (8 waves), tile M=256 (32 rows/wave, 2 A-frags) x N=64, K step 32.
// W chunk (64n x 32k) double-buffered in LDS as bf16 (packed b128 stores);
// each B-fragment LDS read feeds two WMMAs. One barrier per K iteration.
// GEMM1 concat handled via Ksplit: k >= Ksplit reads row-independent vector A2.
template <typename TA, typename TC, bool GELU_ACT>
__global__ __launch_bounds__(256)
void gemm_wmma(const TA* __restrict__ A, int ldA, int Ksplit, const float* __restrict__ A2,
               const float* __restrict__ W, const float* __restrict__ bias,
               TC* __restrict__ C, int M, int N, int K)
{
    __shared__ __bf16 ldsB[2][64 * 32];   // [n][k] bf16, 2 x 4 KB
    const int tid  = threadIdx.x;
    const int wave = tid >> 5, lane = tid & 31;
    const int lm = lane & 15, lh = lane >> 4;
    const int m0 = blockIdx.x * 256 + wave * 32;
    const int nb = blockIdx.y * 64;
    const int n_l = tid >> 2;
    const int k_l = (tid & 3) * 8;
    const float* wbase = W + (size_t)(nb + n_l) * K + k_l;

    v8f acc[2][4] = {};

    // prologue: stage chunk 0
    float4 w0 = *(const float4*)(wbase);
    float4 w1 = *(const float4*)(wbase + 4);
    storeW8(&ldsB[0][n_l * 32 + k_l], w0, w1);
    __syncthreads();

    for (int kk = 0; kk < K; kk += 32) {
        const int  buf  = (kk >> 5) & 1;
        const bool more = (kk + 32) < K;
        float4 nw0, nw1;
        if (more) {
            const float* wp = wbase + kk + 32;
            nw0 = *(const float4*)wp;
            nw1 = *(const float4*)(wp + 4);
            __builtin_prefetch(wp + 32, 0, 0);    // global_prefetch_b8, chunk k+2
        }

        // two A fragments: rows m0+lm and m0+16+lm
        v16bf af0 = cat16(loadA8(A, A2, (size_t)(m0 + lm),      ldA, Ksplit, kk + lh * 8),
                          loadA8(A, A2, (size_t)(m0 + lm),      ldA, Ksplit, kk + 16 + lh * 8));
        v16bf af1 = cat16(loadA8(A, A2, (size_t)(m0 + 16 + lm), ldA, Ksplit, kk + lh * 8),
                          loadA8(A, A2, (size_t)(m0 + 16 + lm), ldA, Ksplit, kk + 16 + lh * 8));

#pragma unroll
        for (int nt = 0; nt < 4; ++nt) {
            const __bf16* bp = &ldsB[buf][(nt * 16 + lm) * 32 + lh * 8];
            v16bf bfrag = cat16(*(const v8bf*)bp, *(const v8bf*)(bp + 16));
            acc[0][nt] = wmma_bf16(af0, bfrag, acc[0][nt]);
            acc[1][nt] = wmma_bf16(af1, bfrag, acc[1][nt]);
        }
        if (more) {
            storeW8(&ldsB[buf ^ 1][n_l * 32 + k_l], nw0, nw1);
            __syncthreads();
        }
    }

    // epilogue: C layout M = lh*8 + r, N = lm (per 16x16 tile)
#pragma unroll
    for (int nt = 0; nt < 4; ++nt) {
        const int col = nb + nt * 16 + lm;
        const float bv = bias ? bias[col] : 0.0f;
#pragma unroll
        for (int ms = 0; ms < 2; ++ms) {
#pragma unroll
            for (int r = 0; r < 8; ++r) {
                const int row = m0 + ms * 16 + lh * 8 + r;
                float v = acc[ms][nt][r] + bv;
                if (GELU_ACT) v = 0.5f * v * (1.0f + erff(v * 0.70710678f));
                C[(size_t)row * N + col] = (TC)v;
            }
        }
    }
}

// ---------------- slot mean: mean_s(slot_init[s,d]*slot_scale[s]) ----------------
__global__ void slot_mean_kernel(const float* __restrict__ slot_init,
                                 const float* __restrict__ slot_scale,
                                 float* __restrict__ mean)
{
    int d = blockIdx.x * 256 + threadIdx.x;
    float acc = 0.f;
#pragma unroll 8
    for (int s = 0; s < SS; ++s) acc += slot_init[s * DM + d] * slot_scale[s];
    mean[d] = acc * (1.0f / SS);
}

// ---------------- top-8 + softmax -> dense alpha; one wave per token ----------------
__global__ __launch_bounds__(256)
void topk_kernel(const float* __restrict__ logits, const float* __restrict__ tau,
                 float* __restrict__ alpha)
{
    const int token = blockIdx.x * 8 + (threadIdx.x >> 5);
    const int lane  = threadIdx.x & 31;
    const float inv = 1.0f / (fabsf(tau[0]) + 0.1f);
    const float* lp = logits + (size_t)token * SS;
    float v0 = lp[lane] * inv;
    float v1 = lp[lane + 32] * inv;

    float selv[TOPK]; int seli[TOPK];
#pragma unroll
    for (int j = 0; j < TOPK; ++j) {
        float bv; int bi;
        if (v0 >= v1) { bv = v0; bi = lane; } else { bv = v1; bi = lane + 32; }
#pragma unroll
        for (int off = 16; off; off >>= 1) {
            float ov = __shfl_xor(bv, off, 32);
            int   oi = __shfl_xor(bi, off, 32);
            if (ov > bv || (ov == bv && oi < bi)) { bv = ov; bi = oi; }
        }
        selv[j] = bv; seli[j] = bi;
        if (bi == lane)            v0 = -3.0e38f;
        else if (bi == lane + 32)  v1 = -3.0e38f;
    }
    const float m = selv[0];
    float sum = 0.f;
#pragma unroll
    for (int j = 0; j < TOPK; ++j) { selv[j] = __expf(selv[j] - m); sum += selv[j]; }
    const float rs = 1.0f / sum;
    float w0 = 0.f, w1 = 0.f;
#pragma unroll
    for (int j = 0; j < TOPK; ++j) {
        float w = selv[j] * rs;
        if (seli[j] == lane)      w0 = w;
        if (seli[j] == lane + 32) w1 = w;
    }
    alpha[(size_t)token * SS + lane]      = w0;
    alpha[(size_t)token * SS + lane + 32] = w1;
}

// ---------------- slot weight sums: wsum[b,s] = sum_t alpha (block-parallel reduce) --------
__global__ __launch_bounds__(256)
void wsum_kernel(const float* __restrict__ alpha, float* __restrict__ wsum)
{
    const int row = blockIdx.x;                // b*64+s
    const int b = row >> 6, s = row & 63;
    const int tid = threadIdx.x, wave = tid >> 5, lane = tid & 31;
    const float* p = alpha + (size_t)b * TT * SS + s;
    float acc = 0.f;
    for (int t = tid; t < TT; t += 256) acc += p[(size_t)t * SS];
#pragma unroll
    for (int off = 16; off; off >>= 1) acc += __shfl_xor(acc, off, 32);
    __shared__ float red[8];
    if (lane == 0) red[wave] = acc;
    __syncthreads();
    if (tid == 0) {
        float t = 0.f;
#pragma unroll
        for (int j = 0; j < 8; ++j) t += red[j];
        wsum[row] = t;
    }
}

// ---------------- dispatch: slot_input[b,s,:] = sum_t alpha[b,t,s] * x[b,t,:] ----------------
// per-block: batch b, 64-wide d tile. M=64(s) x N=64(d), K=T.
// Both operands transposed into double-buffered LDS; one barrier per K chunk.
__global__ __launch_bounds__(256)
void dispatch_wmma(const float* __restrict__ alpha, const float* __restrict__ x,
                   float* __restrict__ slot_input)
{
    __shared__ __bf16 lA[2][64 * 32];   // [s][t]
    __shared__ __bf16 lB[2][64 * 32];   // [d][t]
    const int b = blockIdx.x, db = blockIdx.y * 64;
    const int tid = threadIdx.x, wave = tid >> 5, lane = tid & 31;
    const int lm = lane & 15, lh = lane >> 4;
    const int m_sub = wave & 3;        // s tile
    const int ng = wave >> 2;          // d tile pair
    const int tl = tid >> 3;           // 0..31 (t within chunk)
    const int c0 = (tid & 7) * 8;      // 8-wide column group
    v8f acc[2] = {};

    const float* abase = alpha + ((size_t)b * TT + tl) * SS + c0;
    const float* xbase = x + ((size_t)b * TT + tl) * DM + db + c0;

    // prologue: stage chunk 0
    float4 a0 = *(const float4*)abase, a1 = *(const float4*)(abase + 4);
    float4 b0 = *(const float4*)xbase, b1 = *(const float4*)(xbase + 4);
    {
        lA[0][(c0+0)*32+tl]=(__bf16)a0.x; lA[0][(c0+1)*32+tl]=(__bf16)a0.y;
        lA[0][(c0+2)*32+tl]=(__bf16)a0.z; lA[0][(c0+3)*32+tl]=(__bf16)a0.w;
        lA[0][(c0+4)*32+tl]=(__bf16)a1.x; lA[0][(c0+5)*32+tl]=(__bf16)a1.y;
        lA[0][(c0+6)*32+tl]=(__bf16)a1.z; lA[0][(c0+7)*32+tl]=(__bf16)a1.w;
        lB[0][(c0+0)*32+tl]=(__bf16)b0.x; lB[0][(c0+1)*32+tl]=(__bf16)b0.y;
        lB[0][(c0+2)*32+tl]=(__bf16)b0.z; lB[0][(c0+3)*32+tl]=(__bf16)b0.w;
        lB[0][(c0+4)*32+tl]=(__bf16)b1.x; lB[0][(c0+5)*32+tl]=(__bf16)b1.y;
        lB[0][(c0+6)*32+tl]=(__bf16)b1.z; lB[0][(c0+7)*32+tl]=(__bf16)b1.w;
    }
    __syncthreads();

    for (int t0 = 0; t0 < TT; t0 += 32) {
        const int  buf  = (t0 >> 5) & 1;
        const bool more = (t0 + 32) < TT;
        if (more) {
            a0 = *(const float4*)(abase + (size_t)(t0 + 32) * SS);
            a1 = *(const float4*)(abase + (size_t)(t0 + 32) * SS + 4);
            b0 = *(const float4*)(xbase + (size_t)(t0 + 32) * DM);
            b1 = *(const float4*)(xbase + (size_t)(t0 + 32) * DM + 4);
        }

        const __bf16* apf = &lA[buf][(m_sub * 16 + lm) * 32];
        v16bf afrag = cat16(*(const v8bf*)(apf + lh * 8), *(const v8bf*)(apf + 16 + lh * 8));
#pragma unroll
        for (int i = 0; i < 2; ++i) {
            const int nt = ng * 2 + i;
            const __bf16* bp = &lB[buf][(nt * 16 + lm) * 32 + lh * 8];
            v16bf bfrag = cat16(*(const v8bf*)bp, *(const v8bf*)(bp + 16));
            acc[i] = wmma_bf16(afrag, bfrag, acc[i]);
        }
        if (more) {
            const int nbuf = buf ^ 1;
            lA[nbuf][(c0+0)*32+tl]=(__bf16)a0.x; lA[nbuf][(c0+1)*32+tl]=(__bf16)a0.y;
            lA[nbuf][(c0+2)*32+tl]=(__bf16)a0.z; lA[nbuf][(c0+3)*32+tl]=(__bf16)a0.w;
            lA[nbuf][(c0+4)*32+tl]=(__bf16)a1.x; lA[nbuf][(c0+5)*32+tl]=(__bf16)a1.y;
            lA[nbuf][(c0+6)*32+tl]=(__bf16)a1.z; lA[nbuf][(c0+7)*32+tl]=(__bf16)a1.w;
            lB[nbuf][(c0+0)*32+tl]=(__bf16)b0.x; lB[nbuf][(c0+1)*32+tl]=(__bf16)b0.y;
            lB[nbuf][(c0+2)*32+tl]=(__bf16)b0.z; lB[nbuf][(c0+3)*32+tl]=(__bf16)b0.w;
            lB[nbuf][(c0+4)*32+tl]=(__bf16)b1.x; lB[nbuf][(c0+5)*32+tl]=(__bf16)b1.y;
            lB[nbuf][(c0+6)*32+tl]=(__bf16)b1.z; lB[nbuf][(c0+7)*32+tl]=(__bf16)b1.w;
            __syncthreads();
        }
    }
#pragma unroll
    for (int i = 0; i < 2; ++i) {
        const int d = db + (ng * 2 + i) * 16 + lm;
#pragma unroll
        for (int r = 0; r < 8; ++r) {
            const int s = m_sub * 16 + lh * 8 + r;
            slot_input[((size_t)b * SS + s) * DM + d] = acc[i][r];
        }
    }
}

// ---------------- GRU prep: gi = slot_input/(wsum+1e-8); hs = slot_init*slot_scale ----------------
__global__ void prep_gru(const float* __restrict__ slot_input, const float* __restrict__ wsum,
                         const float* __restrict__ slot_init, const float* __restrict__ slot_scale,
                         float* __restrict__ gi, float* __restrict__ hs)
{
    int idx = blockIdx.x * 256 + threadIdx.x;     // BB*SS*DM
    int row = idx >> 10, d = idx & (DM - 1), s = row & 63;
    gi[idx] = slot_input[idx] / (wsum[row] + 1e-8f);
    hs[idx] = slot_init[s * DM + d] * slot_scale[s];
}

// ---------------- GRU gates ----------------
__global__ void gru_gate(const float* __restrict__ gx, const float* __restrict__ gh,
                         const float* __restrict__ hs, float* __restrict__ snew)
{
    int idx = blockIdx.x * 256 + threadIdx.x;
    int row = idx >> 10, d = idx & (DM - 1);
    const float* gxr = gx + (size_t)row * (3 * DM);
    const float* ghr = gh + (size_t)row * (3 * DM);
    float r = 1.0f / (1.0f + __expf(-(gxr[d] + ghr[d])));
    float z = 1.0f / (1.0f + __expf(-(gxr[DM + d] + ghr[DM + d])));
    float n = tanhf(gxr[2 * DM + d] + r * ghr[2 * DM + d]);
    snew[idx] = (1.0f - z) * n + z * hs[idx];
}

// ---------------- combine: comb[b,t,:] = sum_s alpha[b,t,s]*sv[b,s,:]  (bf16 out) ----------------
__global__ __launch_bounds__(256)
void combine_wmma(const float* __restrict__ alpha, const float* __restrict__ sv,
                  __bf16* __restrict__ comb)
{
    __shared__ __bf16 lB[64 * 64];   // [d][s]
    const int tt = blockIdx.x, db = blockIdx.y * 64, b = blockIdx.z;
    const int tid = threadIdx.x, wave = tid >> 5, lane = tid & 31;
    const int lm = lane & 15, lh = lane >> 4;
    const int m0 = tt * 128 + wave * 16;

    {   // stage sv tile transposed
        const int s = tid >> 2, d0 = (tid & 3) * 16;
        const float* p = sv + ((size_t)b * SS + s) * DM + db + d0;
#pragma unroll
        for (int j = 0; j < 16; j += 4) {
            float4 q = *(const float4*)(p + j);
            lB[(d0+j+0)*64+s]=(__bf16)q.x; lB[(d0+j+1)*64+s]=(__bf16)q.y;
            lB[(d0+j+2)*64+s]=(__bf16)q.z; lB[(d0+j+3)*64+s]=(__bf16)q.w;
        }
    }
    __syncthreads();

    v8f acc[4] = {};
#pragma unroll
    for (int kk = 0; kk < SS; kk += 32) {
        const float* ap = alpha + ((size_t)b * TT + m0 + lm) * SS + kk;
        v16bf afrag = cat16(cvt8(ap + lh * 8), cvt8(ap + 16 + lh * 8));
#pragma unroll
        for (int nt = 0; nt < 4; ++nt) {
            const __bf16* bp = &lB[(nt * 16 + lm) * 64 + kk + lh * 8];
            v16bf bfrag = cat16(*(const v8bf*)bp, *(const v8bf*)(bp + 16));
            acc[nt] = wmma_bf16(afrag, bfrag, acc[nt]);
        }
    }
#pragma unroll
    for (int nt = 0; nt < 4; ++nt) {
        const int d = db + nt * 16 + lm;
#pragma unroll
        for (int r = 0; r < 8; ++r) {
            const int row = m0 + lh * 8 + r;
            comb[((size_t)b * TT + row) * DM + d] = (__bf16)acc[nt][r];
        }
    }
}

// ---------------- host-side orchestration ----------------
extern "C" void kernel_launch(void* const* d_in, const int* in_sizes, int n_in,
                              void* d_out, int out_size, void* d_ws, size_t ws_size,
                              hipStream_t stream)
{
    const float* x          = (const float*)d_in[0];
    const float* slot_init  = (const float*)d_in[1];
    const float* slot_scale = (const float*)d_in[2];
    const float* w1         = (const float*)d_in[3];
    const float* b1         = (const float*)d_in[4];
    const float* w2         = (const float*)d_in[5];
    const float* b2         = (const float*)d_in[6];
    const float* gru_wih    = (const float*)d_in[7];
    const float* gru_whh    = (const float*)d_in[8];
    const float* gru_bih    = (const float*)d_in[9];
    const float* gru_bhh    = (const float*)d_in[10];
    const float* wv         = (const float*)d_in[11];
    const float* bv         = (const float*)d_in[12];
    const float* wo         = (const float*)d_in[13];
    const float* bo         = (const float*)d_in[14];
    const float* tau        = (const float*)d_in[15];
    float* out              = (float*)d_out;

    // workspace layout (fp32 region then bf16 region, 16B aligned throughout)
    float* wsf        = (float*)d_ws;
    float* slot_mean  = wsf;                               // 1024
    float* logits     = slot_mean + DM;                    // BT*SS
    float* alpha      = logits + (size_t)BT * SS;          // BT*SS
    float* slot_input = alpha + (size_t)BT * SS;           // 256*DM
    float* wsum       = slot_input + (size_t)BB*SS*DM;     // 256
    float* gi         = wsum + 256;                        // 256*DM
    float* hs         = gi + (size_t)BB*SS*DM;             // 256*DM
    float* gx         = hs + (size_t)BB*SS*DM;             // 256*3DM
    float* gh         = gx + (size_t)BB*SS*3*DM;           // 256*3DM
    float* snew       = gh + (size_t)BB*SS*3*DM;           // 256*DM
    float* sv         = snew + (size_t)BB*SS*DM;           // 256*DM
    __bf16* h_bf      = (__bf16*)(sv + (size_t)BB*SS*DM);  // BT*HH bf16
    __bf16* comb      = h_bf + (size_t)BT * HH;            // BT*DM bf16

    // 1) slot mean vector (batch-independent)
    slot_mean_kernel<<<DM / 256, 256, 0, stream>>>(slot_init, slot_scale, slot_mean);

    // 2) h = gelu([x | slot_mean] @ w1^T + b1)   -> bf16
    gemm_wmma<float, __bf16, true><<<dim3(BT / 256, HH / 64), 256, 0, stream>>>(
        x, DM, DM, slot_mean, w1, b1, h_bf, BT, HH, 2 * DM);

    // 3) route logits = h @ w2^T + b2
    gemm_wmma<__bf16, float, false><<<dim3(BT / 256, SS / 64), 256, 0, stream>>>(
        h_bf, HH, HH, nullptr, w2, b2, logits, BT, SS, HH);

    // 4) top-8 + softmax -> dense alpha (tau scaling applied here)
    topk_kernel<<<BT / 8, 256, 0, stream>>>(logits, tau, alpha);

    // 5) slot weight sums
    wsum_kernel<<<BB * SS, 256, 0, stream>>>(alpha, wsum);

    // 6) dispatch: slot_input = alpha^T @ x
    dispatch_wmma<<<dim3(BB, DM / 64), 256, 0, stream>>>(alpha, x, slot_input);

    // 7) gi = slot_input/wsum ; hs = slot_init*slot_scale (broadcast over batch)
    prep_gru<<<(BB * SS * DM) / 256, 256, 0, stream>>>(slot_input, wsum, slot_init, slot_scale, gi, hs);

    // 8) GRU matmuls: gx = gi @ wih^T + bih ; gh = hs @ whh^T + bhh
    gemm_wmma<float, float, false><<<dim3((BB*SS) / 256, (3*DM) / 64), 256, 0, stream>>>(
        gi, DM, DM, nullptr, gru_wih, gru_bih, gx, BB * SS, 3 * DM, DM);
    gemm_wmma<float, float, false><<<dim3((BB*SS) / 256, (3*DM) / 64), 256, 0, stream>>>(
        hs, DM, DM, nullptr, gru_whh, gru_bhh, gh, BB * SS, 3 * DM, DM);

    // 9) GRU gates -> new slot states
    gru_gate<<<(BB * SS * DM) / 256, 256, 0, stream>>>(gx, gh, hs, snew);

    // 10) slot_values = S_new @ wv^T + bv
    gemm_wmma<float, float, false><<<dim3((BB*SS) / 256, DM / 64), 256, 0, stream>>>(
        snew, DM, DM, nullptr, wv, bv, sv, BB * SS, DM, DM);

    // 11) combine: comb = alpha @ slot_values  (bf16)
    combine_wmma<<<dim3(TT / 128, DM / 64, BB), 256, 0, stream>>>(alpha, sv, comb);

    // 12) out = comb @ wo^T + bo
    gemm_wmma<__bf16, float, false><<<dim3(BT / 256, DM / 64), 256, 0, stream>>>(
        comb, DM, DM, nullptr, wo, bo, out, BT, DM, DM);
}
